// GCN_20091857010810
// MI455X (gfx1250) — compile-verified, matrix operand order checked by reading, not run on hardware
//
#include <hip/hip_runtime.h>
#include <stdint.h>

#define N_NODES  100000
#define N_EDGES  1600000
#define N_GRAPHS 256
#define D_HID    128
#define D_OUT    64

typedef __attribute__((ext_vector_type(16))) __bf16 v16bf;
typedef __attribute__((ext_vector_type(16))) short  v16s;
typedef __attribute__((ext_vector_type(8)))  short  v8s;
typedef __attribute__((ext_vector_type(8)))  float  v8f;

__device__ __forceinline__ unsigned short f2bf_bits(float f) {
    unsigned int u = __float_as_uint(f);
    u += 0x7FFFu + ((u >> 16) & 1u);          // round-to-nearest-even
    return (unsigned short)(u >> 16);
}

// ---------------- utility ----------------
__global__ void zero_f32(float* __restrict__ p, int n) {
    int i = blockIdx.x * blockDim.x + threadIdx.x;
    if (i < n) p[i] = 0.0f;
}

// ---------------- degree ----------------
__global__ void degree_kernel(const int* __restrict__ dst, float* __restrict__ deg) {
    int e = blockIdx.x * blockDim.x + threadIdx.x;
    if (e < N_EDGES) atomicAdd(&deg[dst[e]], 1.0f);
}

// ---------------- edge scatter: msgs[dst] += h[src] ----------------
// one thread per (edge, 4-channel chunk): float4 gather + 4 f32 atomics (L2-resident)
__global__ void scatter_kernel(const float* __restrict__ h,
                               const int* __restrict__ src,
                               const int* __restrict__ dst,
                               float* __restrict__ msgs) {
    int i = blockIdx.x * blockDim.x + threadIdx.x;
    if (i >= N_EDGES * 32) return;
    int e  = i >> 5;
    int c4 = (i & 31) << 2;
    int s = src[e], d = dst[e];
    float4 v = *(const float4*)(h + (size_t)s * D_HID + c4);
    float* mp = msgs + (size_t)d * D_HID + c4;
    atomicAdd(mp + 0, v.x);
    atomicAdd(mp + 1, v.y);
    atomicAdd(mp + 2, v.z);
    atomicAdd(mp + 3, v.w);
}

// ---------------- weight prep: WcatT[n][k] = bf16([W_rel;W_root][k][n]) ----------------
__global__ void wcat_kernel(const float* __restrict__ Wrel,
                            const float* __restrict__ Wroot,
                            unsigned short* __restrict__ WcatT) {
    int i = blockIdx.x * blockDim.x + threadIdx.x;    // i = n*256 + k
    if (i >= 128 * 256) return;
    int n = i >> 8, k = i & 255;
    float v = (k < 128) ? Wrel[k * 128 + n] : Wroot[(k - 128) * 128 + n];
    WcatT[(size_t)n * 256 + k] = f2bf_bits(v);
}

// ---------------- fused GraphConv GEMM ----------------
// out[m][n] = relu?( (msgs[m]/max(deg,1)) @ W_rel + h[m] @ W_root + b )[n]
// Block: 16 rows x 128 cols. Stage A-panel (16x256, invdeg-scaled, bf16) in LDS once,
// then 8 wave32s each compute one 16x16 tile column via 8 bf16 WMMAs over K=256.
#define A_STRIDE 264   // shorts; 264*2=528B row stride -> rows map to distinct bank groups

__global__ __launch_bounds__(256) void gemm_kernel(
    const float* __restrict__ msgs, const float* __restrict__ hprev,
    const float* __restrict__ deg,  const unsigned short* __restrict__ WcatT,
    const float* __restrict__ bias, float* __restrict__ out, int do_relu)
{
    __shared__ __align__(16) unsigned short Abf[16 * A_STRIDE];

    const int tid  = threadIdx.x;
    const int lane = tid & 31;
    const int wave = tid >> 5;

    // ---- cooperative staging: 16 rows x 256 k, 16 bf16 elems per thread ----
    {
        const int row = tid >> 4;            // 0..15
        const int k0  = (tid & 15) << 4;     // 0,16,...,240
        const int grow = blockIdx.x * 16 + row;
        const float* sp;
        float scale;
        if (k0 < 128) { sp = msgs  + (size_t)grow * 128 + k0;         scale = 1.0f / fmaxf(deg[grow], 1.0f); }
        else          { sp = hprev + (size_t)grow * 128 + (k0 - 128); scale = 1.0f; }
        float fa[16];
        *(float4*)(fa + 0)  = *(const float4*)(sp + 0);
        *(float4*)(fa + 4)  = *(const float4*)(sp + 4);
        *(float4*)(fa + 8)  = *(const float4*)(sp + 8);
        *(float4*)(fa + 12) = *(const float4*)(sp + 12);
        v8s s0, s1;
        #pragma unroll
        for (int i = 0; i < 8; ++i) {
            s0[i] = (short)f2bf_bits(fa[i] * scale);
            s1[i] = (short)f2bf_bits(fa[i + 8] * scale);
        }
        unsigned short* dp = &Abf[row * A_STRIDE + k0];
        *(v8s*)(dp)     = s0;
        *(v8s*)(dp + 8) = s1;
    }
    __syncthreads();

    // ---- per-wave 16x16 tile over K=256 ----
    const int col = wave * 16 + (lane & 15);          // global output column (0..127)
    const int kb  = (lane >> 4) * 8;                  // K sub-group per ISA A/B layout

    const unsigned short* Arow  = &Abf[(lane & 15) * A_STRIDE + kb];
    const unsigned short* Bbase = WcatT + (size_t)col * 256 + kb;

    v8f c = {0.f, 0.f, 0.f, 0.f, 0.f, 0.f, 0.f, 0.f};

    #pragma unroll
    for (int step = 0; step < 8; ++step) {
        // A fragment from LDS: elems 0..7 at k=step*32+kb, elems 8..15 at +16
        const unsigned short* Ap = Arow + step * 32;
        v8s alo = *(const v8s*)(Ap);
        v8s ahi = *(const v8s*)(Ap + 16);
        v16s as = __builtin_shufflevector(alo, ahi,
                    0,1,2,3,4,5,6,7,8,9,10,11,12,13,14,15);

        // B fragment from transposed bf16 weights (L2-hot)
        const unsigned short* Bp = Bbase + step * 32;
        v8s blo = *(const v8s*)(Bp);
        v8s bhi = *(const v8s*)(Bp + 16);
        v16s bs = __builtin_shufflevector(blo, bhi,
                    0,1,2,3,4,5,6,7,8,9,10,11,12,13,14,15);

        union { v16s s; v16bf b; } ua, ub;
        ua.s = as; ub.s = bs;
        c = __builtin_amdgcn_wmma_f32_16x16x32_bf16(
                false, ua.b, false, ub.b, (short)0, c, false, false);
    }

    // C layout: VGPR r -> M = r + 8*(lane>>4), N = lane&15
    const int rbase = blockIdx.x * 16 + 8 * (lane >> 4);
    const float bn = bias[col];
    #pragma unroll
    for (int r = 0; r < 8; ++r) {
        float v = c[r] + bn;
        if (do_relu) v = fmaxf(v, 0.0f);
        out[(size_t)(rbase + r) * 128 + col] = v;
    }
}

// ---------------- global mean pool ----------------
__global__ void pool_kernel(const float* __restrict__ h, const int* __restrict__ batch,
                            float* __restrict__ sums, float* __restrict__ cnt) {
    int i = blockIdx.x * blockDim.x + threadIdx.x;
    if (i >= N_NODES * 32) return;
    int nd = i >> 5;
    int c4 = (i & 31) << 2;
    int g = batch[nd];
    float4 v = *(const float4*)(h + (size_t)nd * 128 + c4);
    float* sp = sums + (size_t)g * 128 + c4;
    atomicAdd(sp + 0, v.x);
    atomicAdd(sp + 1, v.y);
    atomicAdd(sp + 2, v.z);
    atomicAdd(sp + 3, v.w);
    if (c4 == 0) atomicAdd(&cnt[g], 1.0f);
}

// ---------------- head: out = (sums/cnt) @ Wl + bl ----------------
__global__ void head_kernel(const float* __restrict__ sums, const float* __restrict__ cnt,
                            const float* __restrict__ Wl, const float* __restrict__ bl,
                            float* __restrict__ out) {
    int i = blockIdx.x * blockDim.x + threadIdx.x;
    if (i >= N_GRAPHS * D_OUT) return;
    int g = i / D_OUT, o = i % D_OUT;
    float inv = 1.0f / fmaxf(cnt[g], 1.0f);
    float acc = bl[o];
    #pragma unroll 4
    for (int k = 0; k < 128; ++k)
        acc += sums[(size_t)g * 128 + k] * inv * Wl[k * D_OUT + o];
    out[i] = acc;
}

extern "C" void kernel_launch(void* const* d_in, const int* in_sizes, int n_in,
                              void* d_out, int out_size, void* d_ws, size_t ws_size,
                              hipStream_t stream) {
    (void)in_sizes; (void)n_in; (void)out_size; (void)ws_size;

    const float* x    = (const float*)d_in[0];
    const int*   ei   = (const int*)d_in[1];
    const int*   src  = ei;
    const int*   dst  = ei + N_EDGES;
    const int*   batch = (const int*)d_in[2];
    const float* W_rel[3]  = {(const float*)d_in[3], (const float*)d_in[6], (const float*)d_in[9]};
    const float* b_rel[3]  = {(const float*)d_in[4], (const float*)d_in[7], (const float*)d_in[10]};
    const float* W_root[3] = {(const float*)d_in[5], (const float*)d_in[8], (const float*)d_in[11]};
    const float* Wl = (const float*)d_in[12];
    const float* bl = (const float*)d_in[13];

    char* ws = (char*)d_ws;
    size_t off = 0;
    auto carve = [&](size_t bytes) { void* p = ws + off; off += (bytes + 255) & ~(size_t)255; return p; };
    float*          deg   = (float*)carve((size_t)N_NODES * 4);
    float*          msgs  = (float*)carve((size_t)N_NODES * D_HID * 4);
    float*          hA    = (float*)carve((size_t)N_NODES * D_HID * 4);
    float*          hB    = (float*)carve((size_t)N_NODES * D_HID * 4);
    unsigned short* WcatT = (unsigned short*)carve((size_t)128 * 256 * 2);
    float*          sums  = (float*)carve((size_t)N_GRAPHS * D_HID * 4);
    float*          cnt   = (float*)carve((size_t)N_GRAPHS * 4);

    const int T = 256;

    // degree (shared across layers)
    zero_f32<<<(N_NODES + T - 1) / T, T, 0, stream>>>(deg, N_NODES);
    degree_kernel<<<(N_EDGES + T - 1) / T, T, 0, stream>>>(dst, deg);

    const float* hin = x;
    float* houts[3] = {hA, hB, hA};
    for (int l = 0; l < 3; ++l) {
        zero_f32<<<((N_NODES * D_HID) + T - 1) / T, T, 0, stream>>>(msgs, N_NODES * D_HID);
        scatter_kernel<<<((N_EDGES * 32) + T - 1) / T, T, 0, stream>>>(hin, src, dst, msgs);
        wcat_kernel<<<((128 * 256) + T - 1) / T, T, 0, stream>>>(W_rel[l], W_root[l], WcatT);
        gemm_kernel<<<N_NODES / 16, T, 0, stream>>>(msgs, hin, deg, WcatT, b_rel[l],
                                                    houts[l], (l < 2) ? 1 : 0);
        hin = houts[l];
    }

    // mean pool + head
    zero_f32<<<((N_GRAPHS * D_HID) + T - 1) / T, T, 0, stream>>>(sums, N_GRAPHS * D_HID);
    zero_f32<<<(N_GRAPHS + T - 1) / T, T, 0, stream>>>(cnt, N_GRAPHS);
    pool_kernel<<<((N_NODES * 32) + T - 1) / T, T, 0, stream>>>(hin, batch, sums, cnt);
    head_kernel<<<((N_GRAPHS * D_OUT) + T - 1) / T, T, 0, stream>>>(sums, cnt, Wl, bl,
                                                                    (float*)d_out);
}
